// MHA_1_55585466745286
// MI455X (gfx1250) — matrix operand order repaired and checked by reading, not compiled, not run on hardware
//
#include <hip/hip_runtime.h>
#include <cstdint>

// ---------------------------------------------------------------------------
// MHA forward for MI455X (gfx1250, wave32, WMMA bf16 16x16x32, f32 accum)
// Async global->LDS staging (ASYNCcnt), double-buffered tiles,
// WMMA-based softmax row sums, ds_swizzle max reductions, 64-key steps.
// ---------------------------------------------------------------------------

typedef __bf16 bf16_t;
typedef __attribute__((ext_vector_type(16))) __bf16 v16bf;
typedef __attribute__((ext_vector_type(8)))  __bf16 v8bf;
typedef __attribute__((ext_vector_type(8)))  float  v8f;

#define D_MODEL   1024
#define N_HEADS   16
#define HEAD_DIM  64
#define BATCH     2
#define SEQ       2048
#define M_TOTAL   (BATCH * SEQ)          // 4096 tokens
#define QK_SCALE  0.125f                 // 1/sqrt(64)

// ---- CDNA5 async copy: global -> LDS, 16 bytes per lane, ASYNCcnt ---------
static __device__ inline void async_b128(uint32_t lds_addr, uint64_t gaddr) {
    asm volatile("global_load_async_to_lds_b128 %0, %1, off"
                 :: "v"(lds_addr), "v"(gaddr) : "memory");
}
static __device__ inline void wait_async0() {
    asm volatile("s_wait_asynccnt 0x0" ::: "memory");
}

// ---- xor-butterfly max over the 16-lane half (LDS swizzle unit, no memory) -
template <int MASK>
static __device__ inline float xor_max16(float v) {
    constexpr int patt = 0x1f | (MASK << 10);   // and=0x1f, or=0, xor=MASK
    float o = __int_as_float(__builtin_amdgcn_ds_swizzle(__float_as_int(v), patt));
    return fmaxf(v, o);
}
static __device__ inline float half_max(float v) {
    v = xor_max16<1>(v);
    v = xor_max16<2>(v);
    v = xor_max16<4>(v);
    v = xor_max16<8>(v);
    return v;
}

// ---- WMMA fragment loaders (per CDNA5 ISA 7.12.2 register layouts) --------

// A-matrix 16x32 bf16: lane (m = lane&15) holds K-chunks at (lane>>4)*8 and +16
static __device__ inline v16bf load_a_frag(const bf16_t* base, int stride) {
    const int lane = threadIdx.x & 31;
    const bf16_t* p = base + (size_t)(lane & 15) * stride + ((lane >> 4) << 3);
    v8bf lo = *(const v8bf*)p;
    v8bf hi = *(const v8bf*)(p + 16);
    v16bf r;
#pragma unroll
    for (int i = 0; i < 8; ++i) { r[i] = lo[i]; r[i + 8] = hi[i]; }
    return r;
}

// B-matrix 32x16 bf16: lane (n = lane&15) holds column n; K-chunks at
// (lane>>4)*8 and +16 (same order as A)
static __device__ inline v16bf load_b_frag(const bf16_t* base, int stride) {
    const int lane = threadIdx.x & 31;
    const bf16_t* p = base + (size_t)(lane & 15) * stride + ((lane >> 4) << 3);
    v8bf lo = *(const v8bf*)p;
    v8bf hi = *(const v8bf*)(p + 16);
    v16bf r;
#pragma unroll
    for (int i = 0; i < 8; ++i) { r[i] = lo[i]; r[i + 8] = hi[i]; }
    return r;
}

static __device__ inline v8f wmma_bf16(v16bf a, v16bf b, v8f c) {
    // (neg_a, A, neg_b, B, c_mod, C, reuse_a, reuse_b)
    return __builtin_amdgcn_wmma_f32_16x16x32_bf16(false, a, false, b,
                                                   (short)0, c, false, false);
}

// ---- f32 -> bf16 conversion ------------------------------------------------

__global__ __launch_bounds__(256)
void cvt_f32_bf16_kernel(const float* __restrict__ in, bf16_t* __restrict__ out, int n) {
    for (int i = blockIdx.x * blockDim.x + threadIdx.x; i < n;
         i += gridDim.x * blockDim.x)
        out[i] = (bf16_t)in[i];
}

// ---- Projection GEMM: Y[m,n] = sum_k X[m,k] * W[n,k] + b[n] ---------------
// Block = 256 threads (8 waves). Block tile 256(M) x 64(N); wave owns a
// 32x64 strip (2 A-frags, 8 WMMAs per K-step, each LDS B-frag reused twice).
// The 64x32 B tile is shared by all waves: async-copied into double-buffered
// LDS so the copy of step k+1 overlaps the WMMAs of step k. All four B-frags
// are preloaded into distinct registers so the 8 ds_load_b128 issue as one
// batch with partial DScnt waits.
// MODE 0: Q layout [B,H,S,hd] bf16 * QK_SCALE
// MODE 1: K layout [B,H,S,hd] bf16
// MODE 2: Vt layout [B,H,hd,S] bf16 (pre-transposed V)
// MODE 3: plain f32 [M, D_MODEL] (final projection into d_out)
template <int MODE>
__global__ __launch_bounds__(256)
void gemm_kernel(const bf16_t* __restrict__ X, const bf16_t* __restrict__ W,
                 const float* __restrict__ bias, void* __restrict__ outv) {
    __shared__ __attribute__((aligned(16))) bf16_t Bt[2][64 * 40]; // pad stride 40

    const int tid  = threadIdx.x;
    const int lane = tid & 31;
    const int wave = tid >> 5;
    const int m0 = blockIdx.y * 256 + wave * 32;
    const int n0 = blockIdx.x * 64;

    // cooperative copy mapping: 64 rows x 32 halves, 16B per thread
    const int brow = tid >> 2;
    const int boff = (tid & 3) * 8;
    const bf16_t* Wg = W + (size_t)(n0 + brow) * D_MODEL + boff;
    const uint32_t blds[2] = {
        (uint32_t)(size_t)&Bt[0][brow * 40 + boff],
        (uint32_t)(size_t)&Bt[1][brow * 40 + boff]
    };
    const bf16_t* Arow0 = X + (size_t)m0 * D_MODEL;
    const bf16_t* Arow1 = Arow0 + (size_t)16 * D_MODEL;

    async_b128(blds[0], (uint64_t)Wg);
    v16bf a0 = load_a_frag(Arow0, D_MODEL);
    v16bf a1 = load_a_frag(Arow1, D_MODEL);
    v8f acc[2][4] = {};
    wait_async0();
    __syncthreads();

    int cur = 0;
    for (int k = 0; k < D_MODEL; k += 32) {
        const int kn = k + 32;
        if (kn < D_MODEL) async_b128(blds[cur ^ 1], (uint64_t)(Wg + kn));
        const int kp = (kn < D_MODEL) ? kn : 0;
        v16bf a0n = load_a_frag(Arow0 + kp, D_MODEL);
        v16bf a1n = load_a_frag(Arow1 + kp, D_MODEL);
        const bf16_t* Bl = &Bt[cur][0];
        v16bf bfr[4];
#pragma unroll
        for (int c = 0; c < 4; ++c) bfr[c] = load_b_frag(Bl + 16 * c * 40, 40);
#pragma unroll
        for (int c = 0; c < 4; ++c) {
            acc[0][c] = wmma_bf16(a0, bfr[c], acc[0][c]);
            acc[1][c] = wmma_bf16(a1, bfr[c], acc[1][c]);
        }
        wait_async0();
        __syncthreads();
        a0 = a0n;
        a1 = a1n;
        cur ^= 1;
    }

    const int half = lane >> 4;
    const int col  = lane & 15;
#pragma unroll
    for (int u = 0; u < 2; ++u) {
#pragma unroll
        for (int c = 0; c < 4; ++c) {
            const int n = n0 + 16 * c + col;
            const float bn = bias[n];
            const int h = n >> 6, d = n & 63;
#pragma unroll
            for (int r = 0; r < 8; ++r) {
                const int m = m0 + u * 16 + r + 8 * half;
                const int b = m >> 11, s = m & 2047;
                const float v = acc[u][c][r] + bn;
                if constexpr (MODE == 0) {
                    ((bf16_t*)outv)[(((size_t)(b * N_HEADS + h) * SEQ) + s) * HEAD_DIM + d] =
                        (bf16_t)(v * QK_SCALE);
                } else if constexpr (MODE == 1) {
                    ((bf16_t*)outv)[(((size_t)(b * N_HEADS + h) * SEQ) + s) * HEAD_DIM + d] =
                        (bf16_t)v;
                } else if constexpr (MODE == 2) {
                    ((bf16_t*)outv)[(((size_t)(b * N_HEADS + h) * HEAD_DIM) + d) * SEQ + s] =
                        (bf16_t)v;
                } else {
                    ((float*)outv)[(size_t)m * D_MODEL + n] = v;
                }
            }
        }
    }
}

// ---- Flash attention ------------------------------------------------------
// Block = 256 threads (8 waves) sharing one (b,h); wave w owns q rows
// [q0, q0+16). 64-key main step: K tile (64x64) and Vt tile (64x64) are
// async-copied into double-buffered LDS once per block (4 b128/thread/step).
// Each wave per step:
//   8 WMMAs  S = Q*K^T (four 16-key blocks)
//   max via ds_swizzle xor-butterfly — once per 64 keys (amortized 2x)
//   P (16x64) staged through per-wave LDS (same-wave DS ops are in-order)
//   2 WMMAs  rowsum = P x ones (broadcast across each 16-lane half)
//   8 WMMAs  O += P*V
__global__ __launch_bounds__(256)
void flash_attn_kernel(const bf16_t* __restrict__ Q, const bf16_t* __restrict__ K,
                       const bf16_t* __restrict__ Vt, bf16_t* __restrict__ O) {
    __shared__ __attribute__((aligned(16))) bf16_t Kt[2][64 * 72];  // keys x hd
    __shared__ __attribute__((aligned(16))) bf16_t Vl[2][64 * 72];  // hd x keys
    __shared__ __attribute__((aligned(16))) bf16_t Pt[8][16 * 72];  // per-wave P

    const int tid  = threadIdx.x;
    const int lane = tid & 31;
    const int wave = tid >> 5;
    const int gid  = blockIdx.x;
    const int qblk = gid & 15;            // 16 blocks per (b,h)
    const int h    = (gid >> 4) & 15;
    const int b    = gid >> 8;
    const int q0   = (qblk * 8 + wave) * 16;

    const size_t bh = (size_t)(b * N_HEADS + h);
    const bf16_t* Qp  = Q  + (bh * SEQ + q0) * HEAD_DIM;
    const bf16_t* Kbh = K  + bh * SEQ * HEAD_DIM;
    const bf16_t* Vbh = Vt + bh * HEAD_DIM * SEQ;

    // cooperative copy mapping: both tiles are 64 rows x 64 halves (8 KB);
    // each thread copies 2x16B per tile.
    const int crow = tid >> 2;            // 0..63
    const int coff = (tid & 3) * 16;      // halves: 0,16,32,48
    const bf16_t* Kg = Kbh + (size_t)crow * HEAD_DIM + coff;
    const bf16_t* Vg = Vbh + (size_t)crow * SEQ + coff;
    const uint32_t klds[2] = {
        (uint32_t)(size_t)&Kt[0][crow * 72 + coff],
        (uint32_t)(size_t)&Kt[1][crow * 72 + coff]
    };
    const uint32_t vlds[2] = {
        (uint32_t)(size_t)&Vl[0][crow * 72 + coff],
        (uint32_t)(size_t)&Vl[1][crow * 72 + coff]
    };

    // prefetch tile t=0
    async_b128(klds[0],      (uint64_t)Kg);
    async_b128(klds[0] + 16, (uint64_t)(Kg + 8));
    async_b128(vlds[0],      (uint64_t)Vg);
    async_b128(vlds[0] + 16, (uint64_t)(Vg + 8));

    const v16bf qf0 = load_a_frag(Qp,      HEAD_DIM);   // hd 0..31 (pre-scaled)
    const v16bf qf1 = load_a_frag(Qp + 32, HEAD_DIM);   // hd 32..63

    v16bf ones;
#pragma unroll
    for (int i = 0; i < 16; ++i) ones[i] = (bf16_t)1.0f;

    float mrow[8], lrow[8];
#pragma unroll
    for (int r = 0; r < 8; ++r) { mrow[r] = -1e30f; lrow[r] = 0.0f; }
    v8f acc[4] = {};

    bf16_t* Pw = &Pt[wave][0];
    const int half = lane >> 4;
    const int col  = lane & 15;

    wait_async0();
    __syncthreads();

    int cur = 0;
    for (int t = 0; t < SEQ; t += 64) {
        const int tn = t + 64;
        if (tn < SEQ) {
            const uint32_t kl = klds[cur ^ 1], vl = vlds[cur ^ 1];
            const bf16_t* kg = Kg + (size_t)tn * HEAD_DIM;
            const bf16_t* vg = Vg + tn;
            async_b128(kl,      (uint64_t)kg);
            async_b128(kl + 16, (uint64_t)(kg + 8));
            async_b128(vl,      (uint64_t)vg);
            async_b128(vl + 16, (uint64_t)(vg + 8));
        }

        // S = Q * K^T for 64 keys (four 16-key blocks, contract hd in 2 chunks)
        const bf16_t* Kl = &Kt[cur][0];
        v16bf kb[8];
#pragma unroll
        for (int j = 0; j < 4; ++j) {
            kb[2 * j]     = load_b_frag(Kl + 16 * j * 72,      72);
            kb[2 * j + 1] = load_b_frag(Kl + 16 * j * 72 + 32, 72);
        }
        v8f s[4];
#pragma unroll
        for (int j = 0; j < 4; ++j) {
            s[j] = wmma_bf16(qf0, kb[2 * j],     v8f{});
            s[j] = wmma_bf16(qf1, kb[2 * j + 1], s[j]);
        }

        // running max + exp (row = r + 8*half, spread over 16 lanes)
        float corr[8];
#pragma unroll
        for (int r = 0; r < 8; ++r) {
            const float mx = half_max(fmaxf(fmaxf(s[0][r], s[1][r]),
                                            fmaxf(s[2][r], s[3][r])));
            const float nm = fmaxf(mrow[r], mx);
            corr[r] = __expf(mrow[r] - nm);
            mrow[r] = nm;
#pragma unroll
            for (int j = 0; j < 4; ++j)
                s[j][r] = __expf(s[j][r] - nm);
        }
#pragma unroll
        for (int c = 0; c < 4; ++c)
#pragma unroll
            for (int r = 0; r < 8; ++r)
                acc[c][r] *= corr[r];

        // P (16x64): C/D layout -> per-wave LDS (same-wave DS ops in-order)
#pragma unroll
        for (int r = 0; r < 8; ++r) {
            const int row = r + 8 * half;
#pragma unroll
            for (int j = 0; j < 4; ++j)
                Pw[row * 72 + 16 * j + col] = (bf16_t)s[j][r];
        }
        const v16bf pf0 = load_a_frag(Pw,      72);   // keys 0..31
        const v16bf pf1 = load_a_frag(Pw + 32, 72);   // keys 32..63

        // rowsum(P) via WMMA with all-ones B (broadcast across lane halves)
        v8f sums = wmma_bf16(pf0, ones, v8f{});
        sums = wmma_bf16(pf1, ones, sums);
#pragma unroll
        for (int r = 0; r < 8; ++r)
            lrow[r] = lrow[r] * corr[r] + sums[r];

        // O += P * V (contract 64 keys in two 32-chunks)
        const bf16_t* Vc = &Vl[cur][0];
        v16bf vfr[8];
#pragma unroll
        for (int c = 0; c < 4; ++c) {
            vfr[2 * c]     = load_b_frag(Vc + 16 * c * 72,      72);
            vfr[2 * c + 1] = load_b_frag(Vc + 16 * c * 72 + 32, 72);
        }
#pragma unroll
        for (int c = 0; c < 4; ++c) {
            acc[c] = wmma_bf16(pf0, vfr[2 * c],     acc[c]);
            acc[c] = wmma_bf16(pf1, vfr[2 * c + 1], acc[c]);
        }

        wait_async0();
        __syncthreads();
        cur ^= 1;
    }

    // normalize and store attn output as [B, S, H*hd] bf16
    float rinv[8];
#pragma unroll
    for (int r = 0; r < 8; ++r) rinv[r] = 1.0f / lrow[r];
#pragma unroll
    for (int c = 0; c < 4; ++c) {
#pragma unroll
        for (int r = 0; r < 8; ++r) {
            const int s = q0 + r + 8 * half;
            O[((size_t)(b * SEQ + s)) * D_MODEL + h * HEAD_DIM + c * 16 + col] =
                (bf16_t)(acc[c][r] * rinv[r]);
        }
    }
}

// ---------------------------------------------------------------------------

extern "C" void kernel_launch(void* const* d_in, const int* in_sizes, int n_in,
                              void* d_out, int out_size, void* d_ws, size_t ws_size,
                              hipStream_t stream) {
    const float* x  = (const float*)d_in[0];
    const float* Wq = (const float*)d_in[1];
    const float* bq = (const float*)d_in[2];
    const float* Wk = (const float*)d_in[3];
    const float* bk = (const float*)d_in[4];
    const float* Wv = (const float*)d_in[5];
    const float* bv = (const float*)d_in[6];
    const float* Wo = (const float*)d_in[7];
    const float* bo = (const float*)d_in[8];

    // workspace layout (bf16 elements), total 48 MB — lives in the 192MB L2
    const size_t NX = (size_t)M_TOTAL * D_MODEL;     // 4096*1024
    const size_t NW = (size_t)D_MODEL * D_MODEL;     // 1024*1024
    bf16_t* xb  = (bf16_t*)d_ws;
    bf16_t* wqb = xb  + NX;
    bf16_t* wkb = wqb + NW;
    bf16_t* wvb = wkb + NW;
    bf16_t* wob = wvb + NW;
    bf16_t* Qb  = wob + NW;          // [B,H,S,hd]
    bf16_t* Kb  = Qb  + NX;
    bf16_t* Vtb = Kb  + NX;          // [B,H,hd,S]
    bf16_t* Ob  = Vtb + NX;          // [B,S,D]

    // 1) downconvert activations + weights to bf16
    cvt_f32_bf16_kernel<<<2048, 256, 0, stream>>>(x,  xb,  (int)NX);
    cvt_f32_bf16_kernel<<<1024, 256, 0, stream>>>(Wq, wqb, (int)NW);
    cvt_f32_bf16_kernel<<<1024, 256, 0, stream>>>(Wk, wkb, (int)NW);
    cvt_f32_bf16_kernel<<<1024, 256, 0, stream>>>(Wv, wvb, (int)NW);
    cvt_f32_bf16_kernel<<<1024, 256, 0, stream>>>(Wo, wob, (int)NW);

    // 2) QKV projections (Q pre-scaled, V pre-transposed)
    dim3 gdim(D_MODEL / 64, M_TOTAL / 256);
    gemm_kernel<0><<<gdim, 256, 0, stream>>>(xb, wqb, bq, Qb);
    gemm_kernel<1><<<gdim, 256, 0, stream>>>(xb, wkb, bk, Kb);
    gemm_kernel<2><<<gdim, 256, 0, stream>>>(xb, wvb, bv, Vtb);

    // 3) flash attention: 8 waves/block, one 16-row q tile each
    flash_attn_kernel<<<BATCH * N_HEADS * (SEQ / 128), 256, 0, stream>>>(Qb, Kb, Vtb, Ob);

    // 4) output projection (f32 into d_out)
    gemm_kernel<3><<<gdim, 256, 0, stream>>>(Ob, wob, bo, (void*)d_out);
}
